// BasicMessagePassingNetwork_89103391523366
// MI455X (gfx1250) — matrix operand order repaired and checked by reading, not compiled, not run on hardware
//
#include <hip/hip_runtime.h>
#include <hip/hip_bf16.h>
#include <math.h>

typedef __attribute__((ext_vector_type(16))) _Float16 v16h;
typedef __attribute__((ext_vector_type(8)))  float    v8f;

#define N_NODES_C 100000
#define N_EDGES_C 6400000
#define NF 20

union V16H { v16h v; _Float16 h[16]; };

// -----------------------------------------------------------------------------
// Build one B-matrix tile (32x16, f16) for V_WMMA_F32_16X16X32_F16 from a
// row-major weight matrix W[Kdim x 20].  B layout (wave32):
//   lanes 0-15 : column N = lane,    K = 0..7, 16..23  (16 halves)
//   lanes 16-31: column N = lane-16, K = 8..15, 24..31
// Out-of-range K / columns >= 20 are zero-padded.
// -----------------------------------------------------------------------------
__device__ __forceinline__ v16h load_btile(const float* __restrict__ W,
                                           int Kdim, int colBase, int lane) {
  V16H u;
  const int nl    = lane & 15;
  const int col   = colBase + nl;
  const int kbase = (lane < 16) ? 0 : 8;
#pragma unroll
  for (int i = 0; i < 16; ++i) {
    const int k = kbase + i + ((i >= 8) ? 8 : 0);
    float w = (col < NF && k < Kdim) ? W[k * NF + col] : 0.0f;
    u.h[i] = (_Float16)w;
  }
  return u.v;
}

// -----------------------------------------------------------------------------
// Shared MLP body for one 16-edge tile:
//   m1 = relu(A*Wa + ba); m2 = m1*Wb + bb; atomic-scatter m2 into sum[dst].
// A is in WMMA A-layout (16x32 f16).  The m1 C-layout -> A-layout transpose
// goes through a per-wave LDS tile (16 rows x 32 halves).
// -----------------------------------------------------------------------------
__device__ __forceinline__ void mlp16_scatter(
    v16h a, v16h wa0, v16h wa1, v16h wb0, v16h wb1,
    float ba0, float ba1, float bb0, float bb1,
    _Float16* __restrict__ lds, const int* __restrict__ dstp,
    int ebase, float* __restrict__ sum, int lane)
{
  const int nl    = lane & 15;
  const int mbase = (lane < 16) ? 0 : 8;

  v8f c0 = {}; v8f c1 = {};
  c0 = __builtin_amdgcn_wmma_f32_16x16x32_f16(false, a, false, wa0, (short)0, c0, false, false);
  c1 = __builtin_amdgcn_wmma_f32_16x16x32_f16(false, a, false, wa1, (short)0, c1, false, false);

  // bias + relu, write m1 (16x20, zero-padded to 32 cols) to LDS for transpose.
  // C layout: VGPR r holds row M = mbase + r, column N = nl (tile0) / 16+nl (tile1)
#pragma unroll
  for (int r = 0; r < 8; ++r) {
    float v0 = fmaxf(c0[r] + ba0, 0.0f);
    lds[(mbase + r) * 32 + nl] = (_Float16)v0;
    float v1 = (nl < 4) ? fmaxf(c1[r] + ba1, 0.0f) : 0.0f;  // cols 20..31 -> 0
    lds[(mbase + r) * 32 + 16 + nl] = (_Float16)v1;
  }

  // Re-read m1 in A layout: lane<16 row M=nl K{0..7,16..23}; lane>=16 K{8..15}
  V16H a2;
  if (lane < 16) {
#pragma unroll
    for (int i = 0; i < 8; ++i) {
      a2.h[i]     = lds[nl * 32 + i];
      a2.h[8 + i] = lds[nl * 32 + 16 + i];
    }
  } else {
#pragma unroll
    for (int i = 0; i < 8; ++i) {
      a2.h[i]     = lds[nl * 32 + 8 + i];
      a2.h[8 + i] = (_Float16)0.0f;      // K = 24..31
    }
  }

  v8f d0 = {}; v8f d1 = {};
  d0 = __builtin_amdgcn_wmma_f32_16x16x32_f16(false, a2.v, false, wb0, (short)0, d0, false, false);
  d1 = __builtin_amdgcn_wmma_f32_16x16x32_f16(false, a2.v, false, wb1, (short)0, d1, false, false);

  // scatter-add m2 into per-destination sums
#pragma unroll
  for (int r = 0; r < 8; ++r) {
    const int drow = dstp[ebase + mbase + r];
    atomicAdd(sum + (size_t)drow * NF + nl, d0[r] + bb0);
    if (nl < 4)
      atomicAdd(sum + (size_t)drow * NF + 16 + nl, d1[r] + bb1);
  }
}

// -----------------------------------------------------------------------------
// Layer 1 edge kernel: inputs are x[src] (4) ++ edge_attr (3)  -> K = 0..6
// -----------------------------------------------------------------------------
__global__ __launch_bounds__(256) void edge_layer1(
    const float* __restrict__ x, const int* __restrict__ eidx,
    const float* __restrict__ ea,
    const float* __restrict__ Wa, const float* __restrict__ ba,
    const float* __restrict__ Wb, const float* __restrict__ bb,
    float* __restrict__ sum, int nTiles)
{
  __shared__ _Float16 smem[8][16 * 32];
  const int lane = threadIdx.x & 31;
  const int wib  = threadIdx.x >> 5;
  const int gw   = blockIdx.x * 8 + wib;
  const int gws  = gridDim.x * 8;
  const int nl   = lane & 15;
  const int* dstp = eidx + N_EDGES_C;

  const v16h wa0 = load_btile(Wa, 7,  0, lane);
  const v16h wa1 = load_btile(Wa, 7, 16, lane);
  const v16h wb0 = load_btile(Wb, NF,  0, lane);
  const v16h wb1 = load_btile(Wb, NF, 16, lane);
  const float ba0 = ba[nl];
  const float ba1 = (nl < 4) ? ba[16 + nl] : 0.0f;
  const float bb0 = bb[nl];
  const float bb1 = (nl < 4) ? bb[16 + nl] : 0.0f;

  for (int t = gw; t < nTiles; t += gws) {
    const int ebase = t * 16;
    V16H a;
#pragma unroll
    for (int i = 0; i < 16; ++i) a.h[i] = (_Float16)0.0f;
    if (lane < 16) {               // only K=0..6 nonzero -> lanes 16-31 all zero
      const int e = ebase + nl;
      const int s = eidx[e];
      a.h[0] = (_Float16)x[s * 4 + 0];
      a.h[1] = (_Float16)x[s * 4 + 1];
      a.h[2] = (_Float16)x[s * 4 + 2];
      a.h[3] = (_Float16)x[s * 4 + 3];
      a.h[4] = (_Float16)ea[e * 3 + 0];
      a.h[5] = (_Float16)ea[e * 3 + 1];
      a.h[6] = (_Float16)ea[e * 3 + 2];
    }
    mlp16_scatter(a.v, wa0, wa1, wb0, wb1, ba0, ba1, bb0, bb1,
                  smem[wib], dstp, ebase, sum, lane);
  }
}

// -----------------------------------------------------------------------------
// Layer 2 edge kernel: inputs are h1[src] (20, f16) ++ edge_attr (3) -> K=0..22
// -----------------------------------------------------------------------------
__global__ __launch_bounds__(256) void edge_layer2(
    const _Float16* __restrict__ h1, const int* __restrict__ eidx,
    const float* __restrict__ ea,
    const float* __restrict__ Wa, const float* __restrict__ ba,
    const float* __restrict__ Wb, const float* __restrict__ bb,
    float* __restrict__ sum, int nTiles)
{
  __shared__ _Float16 smem[8][16 * 32];
  const int lane = threadIdx.x & 31;
  const int wib  = threadIdx.x >> 5;
  const int gw   = blockIdx.x * 8 + wib;
  const int gws  = gridDim.x * 8;
  const int nl   = lane & 15;
  const int* dstp = eidx + N_EDGES_C;

  const v16h wa0 = load_btile(Wa, 23,  0, lane);
  const v16h wa1 = load_btile(Wa, 23, 16, lane);
  const v16h wb0 = load_btile(Wb, NF,  0, lane);
  const v16h wb1 = load_btile(Wb, NF, 16, lane);
  const float ba0 = ba[nl];
  const float ba1 = (nl < 4) ? ba[16 + nl] : 0.0f;
  const float bb0 = bb[nl];
  const float bb1 = (nl < 4) ? bb[16 + nl] : 0.0f;

  for (int t = gw; t < nTiles; t += gws) {
    const int ebase = t * 16;
    const int e = ebase + nl;
    const int s = eidx[e];
    V16H a;
    if (lane < 16) {
      // K = 0..7
#pragma unroll
      for (int i = 0; i < 8; ++i) a.h[i] = h1[(size_t)s * NF + i];
      // K = 16..19 -> h1[16..19], K = 20..22 -> edge_attr, K = 23 -> 0
#pragma unroll
      for (int i = 0; i < 4; ++i) a.h[8 + i] = h1[(size_t)s * NF + 16 + i];
      a.h[12] = (_Float16)ea[e * 3 + 0];
      a.h[13] = (_Float16)ea[e * 3 + 1];
      a.h[14] = (_Float16)ea[e * 3 + 2];
      a.h[15] = (_Float16)0.0f;
    } else {
      // K = 8..15
#pragma unroll
      for (int i = 0; i < 8; ++i) a.h[i] = h1[(size_t)s * NF + 8 + i];
#pragma unroll
      for (int i = 0; i < 8; ++i) a.h[8 + i] = (_Float16)0.0f;  // K = 24..31
    }
    mlp16_scatter(a.v, wa0, wa1, wb0, wb1, ba0, ba1, bb0, bb1,
                  smem[wib], dstp, ebase, sum, lane);
  }
}

// -----------------------------------------------------------------------------
// In-degree count (shared by both layers)
// -----------------------------------------------------------------------------
__global__ __launch_bounds__(256) void count_kernel(const int* __restrict__ dst,
                                                    float* __restrict__ cnt, int E)
{
  for (int i = blockIdx.x * blockDim.x + threadIdx.x; i < E;
       i += gridDim.x * blockDim.x)
    atomicAdd(cnt + dst[i], 1.0f);
}

// mean + relu + convert to f16 for the layer-2 gather
__global__ __launch_bounds__(256) void finalize1_kernel(
    const float* __restrict__ sum, const float* __restrict__ cnt,
    _Float16* __restrict__ h1, int total)
{
  for (int i = blockIdx.x * blockDim.x + threadIdx.x; i < total;
       i += gridDim.x * blockDim.x) {
    const int n = i / NF;
    const float c = cnt[n];
    float v = (c > 0.0f) ? sum[i] / c : 0.0f;   // scatter-mean (0 if no edges)
    v = fmaxf(v, 0.0f);                         // inter-layer relu
    h1[i] = (_Float16)v;
  }
}

// mean + 20->3 classifier + sigmoid
__global__ __launch_bounds__(256) void finalize2_kernel(
    const float* __restrict__ sum, const float* __restrict__ cnt,
    const float* __restrict__ Wc, const float* __restrict__ bc,
    float* __restrict__ out, int nNodes)
{
  for (int n = blockIdx.x * blockDim.x + threadIdx.x; n < nNodes;
       n += gridDim.x * blockDim.x) {
    const float c = cnt[n];
    const float inv = (c > 0.0f) ? 1.0f / c : 0.0f;
    float h[NF];
#pragma unroll
    for (int f = 0; f < NF; ++f) h[f] = sum[(size_t)n * NF + f] * inv;
#pragma unroll
    for (int j = 0; j < 3; ++j) {
      float acc = bc[j];
#pragma unroll
      for (int f = 0; f < NF; ++f) acc += h[f] * Wc[f * 3 + j];
      out[(size_t)n * 3 + j] = 1.0f / (1.0f + expf(-acc));
    }
  }
}

// -----------------------------------------------------------------------------
extern "C" void kernel_launch(void* const* d_in, const int* in_sizes, int n_in,
                              void* d_out, int out_size, void* d_ws, size_t ws_size,
                              hipStream_t stream)
{
  const float* x   = (const float*)d_in[0];
  const int*   ei  = (const int*)d_in[1];      // edge_index [2, E]
  const float* ea  = (const float*)d_in[2];
  const float* W1a = (const float*)d_in[3];
  const float* b1a = (const float*)d_in[4];
  const float* W1b = (const float*)d_in[5];
  const float* b1b = (const float*)d_in[6];
  const float* W2a = (const float*)d_in[7];
  const float* b2a = (const float*)d_in[8];
  const float* W2b = (const float*)d_in[9];
  const float* b2b = (const float*)d_in[10];
  const float* Wc  = (const float*)d_in[11];
  const float* bc  = (const float*)d_in[12];
  float* out = (float*)d_out;

  char* w = (char*)d_ws;
  float* sum1 = (float*)w;      w += (size_t)N_NODES_C * NF * sizeof(float);
  float* sum2 = (float*)w;      w += (size_t)N_NODES_C * NF * sizeof(float);
  float* cnt  = (float*)w;      w += (size_t)N_NODES_C * sizeof(float);
  _Float16* h1 = (_Float16*)w;  w += (size_t)N_NODES_C * NF * sizeof(_Float16);

  hipMemsetAsync(sum1, 0, (size_t)N_NODES_C * NF * sizeof(float), stream);
  hipMemsetAsync(sum2, 0, (size_t)N_NODES_C * NF * sizeof(float), stream);
  hipMemsetAsync(cnt,  0, (size_t)N_NODES_C * sizeof(float), stream);

  const int nTiles = N_EDGES_C / 16;   // 400000, exact

  count_kernel<<<2048, 256, 0, stream>>>(ei + N_EDGES_C, cnt, N_EDGES_C);

  edge_layer1<<<1024, 256, 0, stream>>>(x, ei, ea, W1a, b1a, W1b, b1b,
                                        sum1, nTiles);

  finalize1_kernel<<<2048, 256, 0, stream>>>(sum1, cnt, h1, N_NODES_C * NF);

  edge_layer2<<<1024, 256, 0, stream>>>(h1, ei, ea, W2a, b2a, W2b, b2b,
                                        sum2, nTiles);

  finalize2_kernel<<<(N_NODES_C + 255) / 256, 256, 0, stream>>>(
      sum2, cnt, Wc, bc, out, N_NODES_C);
}
